// AttentionLayer_14336600834490
// MI455X (gfx1250) — matrix operand order repaired and checked by reading, not compiled, not run on hardware
//
#include <hip/hip_runtime.h>
#include <hip/hip_bf16.h>

// ---------------------------------------------------------------------------
// Additive (Bahdanau) attention for B=64, T=2048, D=512, U=512 on gfx1250.
// Heavy op: v_proj = values @ W2  (68.7 GFLOP) -> bf16 WMMA, f32 accumulate.
// values (268MB) read exactly twice (score pass + context pass) = ~23us HBM
// floor @23.3TB/s; bf16 WMMA compute ~15-27us -> balanced.
// ---------------------------------------------------------------------------

#define B_   64
#define T_   2048
#define D_   512          // DQ == DV == UNITS == 512
#define TT_  64           // t rows per workgroup in score kernel
#define CTX_ELEMS (B_ * D_)          // 32768
#define ATT_ELEMS (B_ * T_)          // 131072

typedef __attribute__((ext_vector_type(16))) __bf16 v16bf;
typedef __attribute__((ext_vector_type(8)))  float  v8f;
typedef __attribute__((ext_vector_type(4)))  float  v4f;
typedef __attribute__((ext_vector_type(4)))  int    v4i;
typedef __attribute__((ext_vector_type(8)))  int    v8i;

// round-to-nearest-even f32 -> bf16 (as raw u16)
static __device__ __forceinline__ unsigned int f2bf(float f) {
  unsigned int u = __float_as_uint(f);
  u += 0x7FFFu + ((u >> 16) & 1u);
  return u >> 16;
}

// Load one 16x16x32 bf16 fragment stored as two 512B blocks:
// block0 = halves e0..7 (16B/lane), block1 = halves e8..15.
static __device__ __forceinline__ v16bf load_frag(const char* p, int lane) {
  v4i lo = *(const v4i*)(p + lane * 16);
  v4i hi = *(const v4i*)(p + 512 + lane * 16);
  v8i w  = __builtin_shufflevector(lo, hi, 0, 1, 2, 3, 4, 5, 6, 7);
  return __builtin_bit_cast(v16bf, w);
}

static __device__ __forceinline__ float fast_tanh(float x) {
  float e2 = __expf(2.0f * x);
  return 1.0f - 2.0f * __builtin_amdgcn_rcpf(e2 + 1.0f); // sat: +/-1 at inf/0
}

// ---------------------------------------------------------------------------
// Kernel 1: qp[b][u] = query[b] . W1[:,u] + b1[u]    (tiny: 34 MFLOP)
// ---------------------------------------------------------------------------
__global__ __launch_bounds__(256) void qp_kernel(const float* __restrict__ query,
                                                 const float* __restrict__ W1,
                                                 const float* __restrict__ b1,
                                                 float* __restrict__ qp) {
  int idx = blockIdx.x * 256 + threadIdx.x;     // 32768 = 64*512
  int b = idx >> 9, u = idx & 511;
  const float* q = query + (b << 9);
  float acc = b1[u];
  for (int k = 0; k < D_; ++k) acc += q[k] * W1[(k << 9) + u];
  qp[idx] = acc;
}

// ---------------------------------------------------------------------------
// Kernel 2: swizzle W2 (f32 row-major [d][u]) into bf16 WMMA B-fragments.
// B element (k,n) of a 32x16 tile lives at lane = (n&15) + 16*(k>>4),
// half-slot e = k&15 (contiguous k per lane half, same order as the A staging).
// Fragment memory layout: (e>>3)*512 + lane*16 + (e&7)*2   (1KB / fragment).
// Frag id f = utile*16 + kchunk;  32 utiles x 16 kchunks = 512KB, L2-resident.
// ---------------------------------------------------------------------------
__global__ __launch_bounds__(512) void w2_swizzle_kernel(const float* __restrict__ W2,
                                                         unsigned short* __restrict__ w2f) {
  int idx = blockIdx.x * 512 + threadIdx.x;     // 262144 elements
  int f = idx >> 9;
  int r = idx & 511;
  int l = r >> 4, e = r & 15;
  int utile = f >> 4, kc = f & 15;
  int u = (utile << 4) + (l & 15);
  int k = (kc << 5) + ((l >> 4) << 4) + e;
  w2f[(f << 9) + ((e >> 3) << 8) + (l << 3) + (e & 7)] =
      (unsigned short)f2bf(W2[(k << 9) + u]);
}

// ---------------------------------------------------------------------------
// Kernel 3 (main): score[b][t] = sum_u tanh(qp[b][u] + b2[u] + (values@W2)[t][u]) * V[u]
// One WG = (b, 64-row t tile), 256 threads = 8 waves.
// LDS: values tile staged as bf16 A-fragments (64KB) + 8x64 partial scores.
// Wave w owns utiles {2w, 2w+1, 16+2w, 16+2w+1}; per B fragment fetched from
// L2 it runs 4 WMMAs (4 t-tiles) -> W2 crosses L2->WGP once per WG.
// ---------------------------------------------------------------------------
__global__ __launch_bounds__(256) void score_kernel(const float* __restrict__ values,
                                                    const unsigned short* __restrict__ w2f,
                                                    const float* __restrict__ qp,
                                                    const float* __restrict__ b2,
                                                    const float* __restrict__ Vw,
                                                    float* __restrict__ scores) {
  extern __shared__ char smem[];                 // [0,65536): A frags, then 2KB partials
  float* scores_s = (float*)(smem + 65536);      // [8 waves][64 rows]

  const int tid = threadIdx.x;
  const int b  = blockIdx.x >> 5;                // 32 t-tiles per batch
  const int t0 = (blockIdx.x & 31) << 6;

  // ---- Stage values[b, t0..t0+63, :] -> bf16 A-fragments in LDS ----------
  // A element (m,k): lane = m + 16*((k>>4)&1), e = k&15  (k contiguous per lane half)
  for (int j = 0; j < 16; ++j) {
    int c   = tid + j * 256;                     // 4096 chunks of 8 floats
    int i   = c >> 6;                            // row in tile
    int d   = (c & 63) << 3;                     // k start (multiple of 8)
    const float* src = values + (((long)(b * T_ + t0 + i)) << 9) + d;
    v4f f0 = *(const v4f*)(src);
    v4f f1 = *(const v4f*)(src + 4);
    v4i pk = { (int)(f2bf(f0[0]) | (f2bf(f0[1]) << 16)),
               (int)(f2bf(f0[2]) | (f2bf(f0[3]) << 16)),
               (int)(f2bf(f1[0]) | (f2bf(f1[1]) << 16)),
               (int)(f2bf(f1[2]) | (f2bf(f1[3]) << 16)) };
    int kc = d >> 5, kk = d & 31;
    int ttile = i >> 4, m = i & 15;
    char* dst = smem + ((ttile * 16 + kc) << 10) + (((kk >> 3) & 1) << 9)
              + ((m + ((kk >> 4) << 4)) << 4);
    *(v4i*)dst = pk;                             // ds_store_b128, conflict-free
  }
  __syncthreads();

  // ---- WMMA main loop ----------------------------------------------------
  const int wid = tid >> 5, lane = tid & 31;
  float sacc[32];                                // [4 ttiles][8 rows] score partials
  #pragma unroll
  for (int i = 0; i < 32; ++i) sacc[i] = 0.0f;

  for (int j = 0; j < 2; ++j) {
    const int ut0 = (j << 4) + (wid << 1);       // two consecutive utiles
    float qpv[2], Vv[2];
    #pragma unroll
    for (int uu = 0; uu < 2; ++uu) {
      int u = ((ut0 + uu) << 4) + (lane & 15);
      qpv[uu] = qp[(b << 9) + u] + b2[u];
      Vv[uu]  = Vw[u];
    }
    v8f cacc[8];                                 // [uu][ttile] accumulators
    #pragma unroll
    for (int i = 0; i < 8; ++i) cacc[i] = (v8f){};

    const char* bb0 = (const char*)w2f + ((long)ut0 << 14);
    #pragma unroll 4
    for (int kc = 0; kc < 16; ++kc) {
      v16bf bf0 = load_frag(bb0 + (kc << 10), lane);            // global (L2-hot)
      v16bf bf1 = load_frag(bb0 + 16384 + (kc << 10), lane);
      #pragma unroll
      for (int tt = 0; tt < 4; ++tt) {
        v16bf af = load_frag(smem + (tt << 14) + (kc << 10), lane); // LDS
        cacc[tt]     = __builtin_amdgcn_wmma_f32_16x16x32_bf16(
            false, af, false, bf0, (short)0, cacc[tt], false, false);
        cacc[4 + tt] = __builtin_amdgcn_wmma_f32_16x16x32_bf16(
            false, af, false, bf1, (short)0, cacc[4 + tt], false, false);
      }
    }
    // tanh(c + qp + b2) * V epilogue (exp/rcp co-execute with XDL)
    #pragma unroll
    for (int uu = 0; uu < 2; ++uu)
      #pragma unroll
      for (int tt = 0; tt < 4; ++tt)
        #pragma unroll
        for (int r = 0; r < 8; ++r)
          sacc[tt * 8 + r] += fast_tanh(cacc[uu * 4 + tt][r] + qpv[uu]) * Vv[uu];
  }

  // ---- cross-lane then cross-wave reduction (deterministic, no atomics) --
  #pragma unroll
  for (int tt = 0; tt < 4; ++tt)
    #pragma unroll
    for (int r = 0; r < 8; ++r) {
      float s = sacc[tt * 8 + r];
      s += __shfl_xor(s, 1, 32);
      s += __shfl_xor(s, 2, 32);
      s += __shfl_xor(s, 4, 32);
      s += __shfl_xor(s, 8, 32);
      if ((lane & 15) == 0)                      // lanes 0 (m=r) and 16 (m=r+8)
        scores_s[wid * 64 + (tt << 4) + r + ((lane >> 4) << 3)] = s;
    }
  __syncthreads();
  if (tid < 64) {
    float s = 0.0f;
    #pragma unroll
    for (int w = 0; w < 8; ++w) s += scores_s[w * 64 + tid];
    scores[(b << 11) + t0 + tid] = s;            // bV omitted: softmax-invariant
  }
}

// ---------------------------------------------------------------------------
// Kernel 4: softmax over T per batch -> attention weights (output region 2)
// ---------------------------------------------------------------------------
__global__ __launch_bounds__(256) void softmax_kernel(const float* __restrict__ scores,
                                                      float* __restrict__ attn) {
  __shared__ float red[256];
  const int b = blockIdx.x, tid = threadIdx.x;
  const float* s = scores + (b << 11);
  float loc[8], m = -1e30f;
  #pragma unroll
  for (int j = 0; j < 8; ++j) { loc[j] = s[tid + (j << 8)]; m = fmaxf(m, loc[j]); }
  red[tid] = m; __syncthreads();
  for (int off = 128; off > 0; off >>= 1) {
    if (tid < off) red[tid] = fmaxf(red[tid], red[tid + off]);
    __syncthreads();
  }
  float M = red[0]; __syncthreads();
  float sum = 0.0f;
  #pragma unroll
  for (int j = 0; j < 8; ++j) { loc[j] = __expf(loc[j] - M); sum += loc[j]; }
  red[tid] = sum; __syncthreads();
  for (int off = 128; off > 0; off >>= 1) {
    if (tid < off) red[tid] += red[tid + off];
    __syncthreads();
  }
  float inv = 1.0f / red[0];
  #pragma unroll
  for (int j = 0; j < 8; ++j) attn[(b << 11) + tid + (j << 8)] = loc[j] * inv;
}

// ---------------------------------------------------------------------------
// Kernel 5: partial context sums (second streaming pass over values).
// 512 WGs = 64 b x 2 d-blocks x 4 t-blocks; partials -> ws (deterministic).
// ---------------------------------------------------------------------------
__global__ __launch_bounds__(256) void ctx_partial_kernel(const float* __restrict__ values,
                                                          const float* __restrict__ attn,
                                                          float* __restrict__ partials) {
  __shared__ float aw[512];
  const int idx = blockIdx.x, tid = threadIdx.x;
  const int b = idx >> 3, sub = idx & 7;
  const int dblk = sub & 1, tblk = sub >> 1;
  const int tstart = tblk << 9;
  aw[tid]       = attn[(b << 11) + tstart + tid];
  aw[tid + 256] = attn[(b << 11) + tstart + tid + 256];
  __syncthreads();
  const int d = (dblk << 8) + tid;
  const float* vp = values + (((long)(b * T_ + tstart)) << 9) + d;
  float acc = 0.0f;
  #pragma unroll 4
  for (int t = 0; t < 512; ++t) acc += aw[t] * vp[(long)t << 9];
  partials[(((tblk << 6) + b) << 9) + d] = acc;
}

__global__ __launch_bounds__(256) void ctx_reduce_kernel(const float* __restrict__ partials,
                                                         float* __restrict__ ctx) {
  int i = blockIdx.x * 256 + threadIdx.x;       // 32768
  ctx[i] = partials[i] + partials[i + 32768] + partials[i + 65536] + partials[i + 98304];
}

// ---------------------------------------------------------------------------
extern "C" void kernel_launch(void* const* d_in, const int* in_sizes, int n_in,
                              void* d_out, int out_size, void* d_ws, size_t ws_size,
                              hipStream_t stream) {
  (void)in_sizes; (void)n_in; (void)out_size; (void)ws_size;
  const float* query  = (const float*)d_in[0];
  const float* values = (const float*)d_in[1];
  const float* W1     = (const float*)d_in[2];
  const float* b1     = (const float*)d_in[3];
  const float* W2     = (const float*)d_in[4];
  const float* b2     = (const float*)d_in[5];
  const float* Vw     = (const float*)d_in[6];
  // d_in[7] = bV: uniform shift of scores, softmax-invariant -> no output effect.

  float* out  = (float*)d_out;
  float* ctx  = out;                 // [64, 512]
  float* attn = out + CTX_ELEMS;     // [64, 2048, 1]

  char* ws = (char*)d_ws;
  float*          qp       = (float*)(ws);                      // 128 KB
  float*          scores   = (float*)(ws + 131072);             // 512 KB
  unsigned short* w2f      = (unsigned short*)(ws + 655360);    // 512 KB
  float*          partials = (float*)(ws + 1179648);            // 512 KB

  qp_kernel<<<128, 256, 0, stream>>>(query, W1, b1, qp);
  w2_swizzle_kernel<<<512, 512, 0, stream>>>(W2, w2f);

  const int smem_bytes = 65536 + 2048;
  hipFuncSetAttribute((const void*)score_kernel,
                      hipFuncAttributeMaxDynamicSharedMemorySize, smem_bytes);
  score_kernel<<<B_ * (T_ / TT_), 256, smem_bytes, stream>>>(values, w2f, qp, b2, Vw, scores);

  softmax_kernel<<<B_, 256, 0, stream>>>(scores, attn);
  ctx_partial_kernel<<<512, 256, 0, stream>>>(values, attn, partials);
  ctx_reduce_kernel<<<128, 256, 0, stream>>>(partials, ctx);
}
